// SpatialEncoder_60301340836410
// MI455X (gfx1250) — matrix operand order repaired and checked by reading, not compiled, not run on hardware
//
#include <hip/hip_runtime.h>
#include <hip/hip_bf16.h>
#include <math.h>

// Problem constants (match reference shapes)
#define BATCH 8
#define NN    2048
#define SS    16
#define CC    32
#define DD    128
#define ETA_C  1.0f
#define BETA_C 0.6f

typedef __attribute__((ext_vector_type(16))) __bf16 v16bf;
typedef __attribute__((ext_vector_type(8)))  float  v8f;

// --- fragment load helpers (addresses are 16B/32B aligned by construction) ---
static __device__ __forceinline__ v16bf ld_frag_2x16B(const __bf16* p0, const __bf16* p1) {
  union { v16bf v; uint4 u[2]; } z;
  z.u[0] = *(const uint4*)p0;
  z.u[1] = *(const uint4*)p1;
  return z.v;
}
static __device__ __forceinline__ v16bf ld_frag_32B(const __bf16* p) {
  union { v16bf v; uint4 u[2]; } z;
  z.u[0] = *(const uint4*)p;
  z.u[1] = *(const uint4*)(p + 8);
  return z.v;
}

// ============================================================================
// Kernel P: precompute Bias[n][m] = M ? ETA*logit(clip(A_stat)) : -inf
//           and bf16 transposed weight matrices WqT/WkT ([d_out][d_in]).
// ============================================================================
__global__ void prep_kernel(const float* __restrict__ A_stat,
                            const int*   __restrict__ M_mask,
                            const float* __restrict__ Wq,
                            const float* __restrict__ Wk,
                            float* __restrict__ Bias,
                            __bf16* __restrict__ WqT,
                            __bf16* __restrict__ WkT) {
  int stride = gridDim.x * blockDim.x;
  for (int i = blockIdx.x * blockDim.x + threadIdx.x; i < NN * NN; i += stride) {
    float p = A_stat[i];
    p = fminf(fmaxf(p, 1e-3f), 1.0f - 1e-3f);
    float bias = ETA_C * logf(p / (1.0f - p));
    Bias[i] = (M_mask[i] != 0) ? bias : -INFINITY;
    if (i < DD * DD) {
      int d = i >> 7, k = i & (DD - 1);
      WqT[i] = (__bf16)Wq[k * DD + d];
      WkT[i] = (__bf16)Wk[k * DD + d];
    }
  }
}

// ============================================================================
// Kernel A: per 16-row tile (1 wave): masked temporal mean pool (fp32 VALU),
//           Hn = x @ W_proj (fp32 VALU), Q/K = Hn @ Wq/Wk via bf16 WMMA.
//           Emits Qb/Kb (bf16 row-major) and HnT (bf16, [b][d][n]).
// ============================================================================
__global__ __launch_bounds__(32) void pool_proj_qk_kernel(
    const float* __restrict__ X, const float* __restrict__ mask,
    const float* __restrict__ W_proj,
    const __bf16* __restrict__ WqT, const __bf16* __restrict__ WkT,
    __bf16* __restrict__ Qb, __bf16* __restrict__ Kb,
    __bf16* __restrict__ HnT) {
  __shared__ float xp[16][CC];
  __shared__ float Hs[16][DD];
  int tile = blockIdx.x;                      // B * N/16 tiles
  int b  = tile / (NN / 16);
  int n0 = (tile % (NN / 16)) * 16;
  int lane = threadIdx.x;

  // masked temporal mean pooling over S
  for (int idx = lane; idx < 16 * CC; idx += 32) {
    int row = idx >> 5, c = idx & 31;
    const float* xptr = X + ((size_t)(b * NN + n0 + row)) * SS * CC + c;
    const float* mptr = mask + ((size_t)(b * NN + n0 + row)) * SS;
    float s = 0.f, ms = 0.f;
    for (int t = 0; t < SS; ++t) { float m = mptr[t]; s += xptr[t * CC] * m; ms += m; }
    xp[row][c] = s / fmaxf(ms, 1.0f);
  }
  __syncthreads();

  // Hn = x @ W_proj (fp32), stash to LDS + bf16 transposed copy to HnT
  for (int j = lane; j < 16 * DD; j += 32) {
    int row = j >> 7, d = j & (DD - 1);
    float acc = 0.f;
    for (int c = 0; c < CC; ++c) acc += xp[row][c] * W_proj[c * DD + d];
    Hs[row][d] = acc;
    HnT[((size_t)(b * DD + d)) * NN + n0 + row] = (__bf16)acc;
  }
  __syncthreads();

  // Build WMMA A-fragments of Hn (16x32 bf16 per K-block)
  int arow  = lane & 15;
  int base8 = (lane < 16) ? 0 : 8;
  int koff  = (lane < 16) ? 0 : 16;
  int rbase = (lane < 16) ? 0 : 8;
  v16bf afrag[4];
  for (int kb = 0; kb < 4; ++kb) {
    v16bf f;
    for (int e = 0; e < 8; ++e) {
      f[e]     = (__bf16)Hs[arow][kb * 32 + base8 + e];
      f[e + 8] = (__bf16)Hs[arow][kb * 32 + base8 + 16 + e];
    }
    afrag[kb] = f;
  }
  int dcol = lane & 15;
  for (int which = 0; which < 2; ++which) {
    const __bf16* WT  = which ? WkT : WqT;
    __bf16*       Out = which ? Kb  : Qb;
    for (int dt = 0; dt < 8; ++dt) {
      v8f acc = {};
      for (int kb = 0; kb < 4; ++kb) {
        const __bf16* wp = WT + (size_t)(dt * 16 + dcol) * DD + kb * 32 + base8;
        v16bf bf = ld_frag_2x16B(wp, wp + 16);
        acc = __builtin_amdgcn_wmma_f32_16x16x32_bf16(false, afrag[kb], false, bf,
                                                      (short)0, acc, false, false);
      }
      for (int r = 0; r < 8; ++r)
        Out[((size_t)(b * NN + n0 + r + rbase)) * DD + dt * 16 + dcol] = (__bf16)acc[r];
    }
  }
}

// ============================================================================
// Kernel C: per (b, 16-row tile), 256 threads (8 waves).
//   LDS holds the full 16x2048 logits/P row-block (128KB).
//   phase1: logits = (Q K^T)/sqrt(D) + Bias  (bf16 WMMA, fp32 accum)
//   phase2: rowmax, Z (softmax pieces)
//   phase3: P = M ? (beta*e/Z + (1-beta)*A_stat)*r_n*r_m : 0 ; rowsum;
//           A_fuse = P/rowsum -> global
//   phase4: H_unnorm = P @ Hn via bf16 WMMA, cross-wave ds_add_f32 reduce,
//           H = H_unnorm / rowsum -> global
// ============================================================================
__global__ __launch_bounds__(256) void attn_fuse_kernel(
    const __bf16* __restrict__ Qb, const __bf16* __restrict__ Kb,
    const __bf16* __restrict__ HnT,
    const float* __restrict__ Bias, const float* __restrict__ A_stat,
    const int* __restrict__ M_mask, const float* __restrict__ node_reliab,
    float* __restrict__ A_fuse_out, float* __restrict__ H_out) {
  extern __shared__ float smem[];
  float* Lb     = smem;                 // 16*2048 logits, then P in-place
  float* Hacc   = Lb + 16 * NN;         // 16*128
  float* red    = Hacc + 16 * DD;       // 16*16
  float* rowmax = red + 256;            // 16
  float* rowZ   = rowmax + 16;          // 16
  float* rowsum = rowZ + 16;            // 16
  float* reln   = rowsum + 16;          // 16

  int tile = blockIdx.x;
  int b  = tile / (NN / 16);
  int n0 = (tile % (NN / 16)) * 16;
  int tid  = threadIdx.x;
  int wv   = tid >> 5, lane = tid & 31;

  if (tid < 16) {
    float r = node_reliab[(size_t)b * NN + n0 + tid];
    reln[tid] = fminf(fmaxf(r, 0.f), 1.f);
  }

  // Q A-fragments for this row tile (each wave keeps its own copy)
  int base8 = (lane < 16) ? 0 : 8;
  int koff  = (lane < 16) ? 0 : 16;
  int rbase = (lane < 16) ? 0 : 8;
  int qrow  = n0 + (lane & 15);
  v16bf qfrag[4];
  for (int kb = 0; kb < 4; ++kb) {
    const __bf16* p0 = Qb + ((size_t)(b * NN + qrow)) * DD + kb * 32 + base8;
    qfrag[kb] = ld_frag_2x16B(p0, p0 + 16);
  }
  const float rscale = 0.08838834764831845f;  // 1/sqrt(128)

  // phase 1: logits tile-block -> LDS (each wave owns 16 m-tiles)
  for (int t = wv * 16; t < wv * 16 + 16; ++t) {
    int mcol = t * 16 + (lane & 15);
    v8f acc = {};
    for (int kb = 0; kb < 4; ++kb) {
      const __bf16* kp = Kb + ((size_t)(b * NN + mcol)) * DD + kb * 32 + base8;
      v16bf kf = ld_frag_2x16B(kp, kp + 16);
      acc = __builtin_amdgcn_wmma_f32_16x16x32_bf16(false, qfrag[kb], false, kf,
                                                    (short)0, acc, false, false);
    }
    for (int r = 0; r < 8; ++r) {
      int n = r + rbase;
      Lb[n * NN + mcol] = acc[r] * rscale + Bias[(size_t)(n0 + n) * NN + mcol];
    }
  }
  __syncthreads();

  // phase 2a: row max
  {
    int row = tid >> 4, sub = tid & 15;
    float mx = -INFINITY;
    for (int m = sub; m < NN; m += 16) mx = fmaxf(mx, Lb[row * NN + m]);
    red[row * 16 + sub] = mx;
  }
  __syncthreads();
  if (tid < 16) {
    float mx = -INFINITY;
    for (int s = 0; s < 16; ++s) mx = fmaxf(mx, red[tid * 16 + s]);
    rowmax[tid] = mx;
  }
  __syncthreads();

  // phase 2b: Z = sum exp(l - rowmax)
  {
    int row = tid >> 4, sub = tid & 15;
    float mx = rowmax[row], z = 0.f;
    for (int m = sub; m < NN; m += 16) z += __expf(Lb[row * NN + m] - mx);
    red[row * 16 + sub] = z;
  }
  __syncthreads();
  if (tid < 16) {
    float z = 0.f;
    for (int s = 0; s < 16; ++s) z += red[tid * 16 + s];
    rowZ[tid] = z;
  }
  __syncthreads();

  // phase 3a: P (unnormalized fused adjacency) in place of logits
  for (int idx = tid; idx < 16 * NN; idx += 256) {
    int row = idx >> 11, m = idx & (NN - 1);
    float e = __expf(Lb[idx] - rowmax[row]);
    size_t gi = (size_t)(n0 + row) * NN + m;
    float rm = node_reliab[(size_t)b * NN + m];
    rm = fminf(fmaxf(rm, 0.f), 1.f);
    float P = (M_mask[gi] != 0)
            ? (BETA_C * e / rowZ[row] + (1.0f - BETA_C) * A_stat[gi]) * reln[row] * rm
            : 0.f;
    Lb[idx] = P;
  }
  __syncthreads();

  // phase 3b: rowsum of P
  {
    int row = tid >> 4, sub = tid & 15;
    float s = 0.f;
    for (int m = sub; m < NN; m += 16) s += Lb[row * NN + m];
    red[row * 16 + sub] = s;
  }
  __syncthreads();
  if (tid < 16) {
    float s = 0.f;
    for (int k = 0; k < 16; ++k) s += red[tid * 16 + k];
    rowsum[tid] = s;
  }
  __syncthreads();

  // phase 3c: A_fuse out (coalesced), zero H accumulator
  for (int idx = tid; idx < 16 * NN; idx += 256) {
    int row = idx >> 11, m = idx & (NN - 1);
    A_fuse_out[((size_t)(b * NN + n0 + row)) * NN + m] = Lb[idx] / (rowsum[row] + 1e-8f);
  }
  for (int idx = tid; idx < 16 * DD; idx += 256) Hacc[idx] = 0.f;
  __syncthreads();

  // phase 4: H_unnorm = P @ Hn ; each wave covers 256 m in 8 super-tiles of 32
  v8f hacc[8] = {};
  int prow = lane & 15;
  for (int st = 0; st < 8; ++st) {
    int ms = wv * 256 + st * 32;
    const float* lp = Lb + prow * NN + ms + base8;
    v16bf pf;
    for (int e = 0; e < 8; ++e) { pf[e] = (__bf16)lp[e]; pf[e + 8] = (__bf16)lp[16 + e]; }
    for (int dt = 0; dt < 8; ++dt) {
      int d = dt * 16 + (lane & 15);
      const __bf16* hp = HnT + ((size_t)(b * DD + d)) * NN + ms + base8;
      v16bf hf = ld_frag_2x16B(hp, hp + 16);
      hacc[dt] = __builtin_amdgcn_wmma_f32_16x16x32_bf16(false, pf, false, hf,
                                                         (short)0, hacc[dt], false, false);
    }
  }
  for (int dt = 0; dt < 8; ++dt)
    for (int r = 0; r < 8; ++r)
      atomicAdd(&Hacc[(r + rbase) * DD + dt * 16 + (lane & 15)], hacc[dt][r]);
  __syncthreads();

  // phase 5: H = H_unnorm / rowsum
  for (int idx = tid; idx < 16 * DD; idx += 256) {
    int n = idx >> 7, d = idx & (DD - 1);
    H_out[((size_t)(b * NN + n0 + n)) * DD + d] = Hacc[idx] / (rowsum[n] + 1e-8f);
  }
}

// ============================================================================
// Kernel D: out = LN2(LN1(relu(H @ W_lin)))  — fp32 for LN accuracy.
// One 128-thread block per (b,n) row.
// ============================================================================
__global__ __launch_bounds__(128) void lin_ln_kernel(
    const float* __restrict__ H, const float* __restrict__ W_lin,
    const float* __restrict__ s1, const float* __restrict__ b1,
    const float* __restrict__ s2, const float* __restrict__ b2,
    float* __restrict__ out) {
  __shared__ float h[DD];
  __shared__ float red[DD];
  int row = blockIdx.x;
  int d = threadIdx.x;
  h[d] = H[(size_t)row * DD + d];
  __syncthreads();
  float y = 0.f;
  for (int i = 0; i < DD; ++i) y += h[i] * W_lin[i * DD + d];
  y = fmaxf(y, 0.f);
  // LN1
  red[d] = y; __syncthreads();
  for (int s = 64; s > 0; s >>= 1) { if (d < s) red[d] += red[d + s]; __syncthreads(); }
  float mu = red[0] / DD; __syncthreads();
  float c = y - mu;
  red[d] = c * c; __syncthreads();
  for (int s = 64; s > 0; s >>= 1) { if (d < s) red[d] += red[d + s]; __syncthreads(); }
  float var = red[0] / DD; __syncthreads();
  float h1 = c * rsqrtf(var + 1e-5f) * s1[d] + b1[d];
  // LN2
  red[d] = h1; __syncthreads();
  for (int s = 64; s > 0; s >>= 1) { if (d < s) red[d] += red[d + s]; __syncthreads(); }
  float mu2 = red[0] / DD; __syncthreads();
  float c2 = h1 - mu2;
  red[d] = c2 * c2; __syncthreads();
  for (int s = 64; s > 0; s >>= 1) { if (d < s) red[d] += red[d + s]; __syncthreads(); }
  float var2 = red[0] / DD; __syncthreads();
  out[(size_t)row * DD + d] = c2 * rsqrtf(var2 + 1e-5f) * s2[d] + b2[d];
}

// ============================================================================
// Launch
// ============================================================================
extern "C" void kernel_launch(void* const* d_in, const int* in_sizes, int n_in,
                              void* d_out, int out_size, void* d_ws, size_t ws_size,
                              hipStream_t stream) {
  const float* X_short     = (const float*)d_in[0];
  const float* mask_short  = (const float*)d_in[1];
  const float* A_stat      = (const float*)d_in[2];
  const int*   M_mask      = (const int*)  d_in[3];
  const float* node_reliab = (const float*)d_in[4];
  const float* W_proj      = (const float*)d_in[5];
  const float* Wq          = (const float*)d_in[6];
  const float* Wk          = (const float*)d_in[7];
  const float* W_lin       = (const float*)d_in[8];
  const float* ln1_scale   = (const float*)d_in[9];
  const float* ln1_bias    = (const float*)d_in[10];
  const float* ln2_scale   = (const float*)d_in[11];
  const float* ln2_bias    = (const float*)d_in[12];

  // workspace layout (all 256B-aligned)
  char* ws = (char*)d_ws;
  float*  Bias = (float*) (ws + 0);              // N*N*4        = 16,777,216
  __bf16* Qb   = (__bf16*)(ws + 16777216);       // B*N*D*2      =  4,194,304
  __bf16* Kb   = (__bf16*)(ws + 20971520);       // B*N*D*2
  __bf16* HnT  = (__bf16*)(ws + 25165824);       // B*D*N*2
  float*  Hbuf = (float*) (ws + 29360128);       // B*N*D*4      =  8,388,608
  __bf16* WqT  = (__bf16*)(ws + 37748736);       // D*D*2
  __bf16* WkT  = (__bf16*)(ws + 37781504);       // D*D*2

  float* out_ptr    = (float*)d_out;                       // (B,N,D)
  float* A_fuse_ptr = (float*)d_out + (size_t)BATCH * NN * DD;  // (B,N,N)

  // 1) prep: Bias + bf16 transposed Wq/Wk
  prep_kernel<<<2048, 256, 0, stream>>>(A_stat, M_mask, Wq, Wk, Bias, WqT, WkT);

  // 2) pooling + projection + Q/K (WMMA)
  pool_proj_qk_kernel<<<BATCH * (NN / 16), 32, 0, stream>>>(
      X_short, mask_short, W_proj, WqT, WkT, Qb, Kb, HnT);

  // 3) attention + fuse + renorm + H (WMMA, 137KB dynamic LDS)
  size_t smemC = (size_t)(16 * NN + 16 * DD + 256 + 64) * sizeof(float);
  (void)hipFuncSetAttribute((const void*)attn_fuse_kernel,
                            hipFuncAttributeMaxDynamicSharedMemorySize, (int)smemC);
  attn_fuse_kernel<<<BATCH * (NN / 16), 256, smemC, stream>>>(
      Qb, Kb, HnT, Bias, A_stat, M_mask, node_reliab, A_fuse_ptr, Hbuf);

  // 4) W_lin + relu + LN1 + LN2
  lin_ln_kernel<<<BATCH * NN, 128, 0, stream>>>(
      Hbuf, W_lin, ln1_scale, ln1_bias, ln2_scale, ln2_bias, out_ptr);
}
